// LayerNormLSTMStack_55508157333865
// MI455X (gfx1250) — matrix operand order repaired and checked by reading, not compiled, not run on hardware
//
#include <hip/hip_runtime.h>

// Problem constants
#define BATCH 64
#define TLEN  1024
#define HDIM  512
#define DDIM  512
#define GATES 2048   // 4*H

typedef __attribute__((ext_vector_type(16))) __bf16 v16bf;
typedef __attribute__((ext_vector_type(8)))  __bf16 v8bf;
typedef __attribute__((ext_vector_type(8)))  float  v8f;

// RNE float->bf16 (used for the one-time weight split)
static __device__ __forceinline__ __bf16 f2bf_rne(float f) {
  union { float f; unsigned u; } v; v.f = f;
  unsigned r = v.u + 0x7FFFu + ((v.u >> 16) & 1u);
  return __builtin_bit_cast(__bf16, (unsigned short)(r >> 16));
}
static __device__ __forceinline__ float bf2f(__bf16 b) {
  unsigned short h = __builtin_bit_cast(unsigned short, b);
  union { unsigned u; float f; } v; v.u = ((unsigned)h) << 16;
  return v.f;
}
// Fast truncation-based hi/lo split for the activation staging path.
// r = f - trunc_bf16(f) is exact in f32; truncating lo adds only ~2^-16 |f|.
static __device__ __forceinline__ void split2(float f, __bf16& hi, __bf16& lo) {
  unsigned u  = __builtin_bit_cast(unsigned, f);
  unsigned uh = u & 0xFFFF0000u;
  hi = __builtin_bit_cast(__bf16, (unsigned short)(u >> 16));
  float r = f - __builtin_bit_cast(float, uh);
  unsigned ur = __builtin_bit_cast(unsigned, r);
  lo = __builtin_bit_cast(__bf16, (unsigned short)(ur >> 16));
}

// Split an f32 array into bf16 hi/lo pair (a ~= hi + lo), RNE (prologue only).
__global__ void __launch_bounds__(256)
split_bf16x2(const float* __restrict__ W, __bf16* __restrict__ hi,
             __bf16* __restrict__ lo, int n) {
  int i = blockIdx.x * 256 + threadIdx.x;
  if (i < n) {
    float f = W[i];
    __bf16 h = f2bf_rne(f);
    hi[i] = h;
    lo[i] = f2bf_rne(f - bf2f(h));
  }
}

__global__ void __launch_bounds__(256)
zero4(float* h0, float* c0, float* h1, float* c1, int n) {
  int i = blockIdx.x * 256 + threadIdx.x;
  if (i < n) { h0[i] = 0.f; c0[i] = 0.f; h1[i] = 0.f; c1[i] = 0.f; }
}

// C[64 x N] = concat(A0[64 x K0], A1[64 x (Ktot-K0)]) * W^T + bias
// W pre-split into bf16 hi/lo, row-major [N x Ktot]. bf16x3 WMMA.
// K-chunk = 64 -> 6 WMMAs per barrier pair.
__global__ void __launch_bounds__(256)
gemm_bf16x3(const float* __restrict__ A0, unsigned long long strideA0, int K0,
            const float* __restrict__ A1, unsigned long long strideA1,
            const __bf16* __restrict__ Whi, const __bf16* __restrict__ Wlo,
            const float* __restrict__ bias,
            float* __restrict__ out, int N, int Ktot)
{
  __shared__ __bf16 Ahi[64][72];   // 72: 144B rows (16B-aligned, conflict-light)
  __shared__ __bf16 Alo[64][72];

  const int tid  = threadIdx.x;
  const int lane = tid & 31;
  const int wid  = tid >> 5;
  const int mt   = wid & 3;        // 4 M-tiles
  const int nt   = wid >> 2;       // 2 N-tiles
  const int n0   = blockIdx.x * 32;

  // staging: each thread loads 16 consecutive k of one row
  const int mrow = tid >> 2;            // 0..63
  const int kcol = (tid & 3) * 16;      // 0,16,32,48

  // A fragment addressing (ISA 16-bit A 16x32 layout)
  const int mloc = lane & 15;
  const int kbA  = (lane >= 16) ? 8  : 0;
  // B fragment addressing (ISA 16-bit B 32x16 layout)
  const int kbB  = (lane >= 16) ? 8  : 0;
  const int nglob = n0 + nt * 16 + (lane & 15);

  v8f acc = {};

  const __bf16* wh_base = Whi + (unsigned long long)nglob * Ktot;
  const __bf16* wl_base = Wlo + (unsigned long long)nglob * Ktot;
  const int am = mt * 16 + mloc;

  for (int kc = 0; kc < Ktot; kc += 64) {
    // ---- stage activation chunk into LDS as hi/lo bf16 ----
    {
      int kg = kc + kcol;
      const float* srow = (kg < K0)
          ? (A0 + (unsigned long long)mrow * strideA0 + kg)
          : (A1 + (unsigned long long)mrow * strideA1 + (kg - K0));
      #pragma unroll
      for (int j = 0; j < 16; ++j) {
        __bf16 h, l;
        split2(srow[j], h, l);
        Ahi[mrow][kcol + j] = h;
        Alo[mrow][kcol + j] = l;
      }
    }
    __syncthreads();

    // ---- two 16x16x32 sub-chunks: 6 WMMAs per barrier pair ----
    #pragma unroll
    for (int ks = 0; ks < 64; ks += 32) {
      union { v16bf v; v8bf h[2]; } ah, al, bh, bl;
      ah.h[0] = *(const v8bf*)&Ahi[am][ks + kbA];
      ah.h[1] = *(const v8bf*)&Ahi[am][ks + kbA + 16];
      al.h[0] = *(const v8bf*)&Alo[am][ks + kbA];
      al.h[1] = *(const v8bf*)&Alo[am][ks + kbA + 16];

      const __bf16* bhp = wh_base + kc + ks + kbB;
      const __bf16* blp = wl_base + kc + ks + kbB;
      bh.h[0] = *(const v8bf*)bhp;  bh.h[1] = *(const v8bf*)(bhp + 16);
      bl.h[0] = *(const v8bf*)blp;  bl.h[1] = *(const v8bf*)(blp + 16);

      // bf16x3: hi*hi + hi*lo + lo*hi (drop lo*lo), f32 accumulate
      acc = __builtin_amdgcn_wmma_f32_16x16x32_bf16(false, ah.v, false, bh.v,
                                                    (short)0, acc, false, false);
      acc = __builtin_amdgcn_wmma_f32_16x16x32_bf16(false, ah.v, false, bl.v,
                                                    (short)0, acc, false, false);
      acc = __builtin_amdgcn_wmma_f32_16x16x32_bf16(false, al.v, false, bh.v,
                                                    (short)0, acc, false, false);
    }
    __syncthreads();
  }

  // ---- epilogue: C layout -> VGPR r holds M=r (+8 for upper lanes), N=lane&15
  const float bn    = bias[nglob];
  const int   mbase = mt * 16 + ((lane >= 16) ? 8 : 0);
  float* op = out + (unsigned long long)mbase * N + nglob;
  #pragma unroll
  for (int r = 0; r < 8; ++r)
    op[(unsigned long long)r * N] = acc[r] + bn;
}

// One block per batch row; 512 threads; fused 4-gate LayerNorm + activations +
// cell update + cell-state LayerNorm. Biased variance (matches torch/jax ref).
__global__ void __launch_bounds__(512)
cell_update(const float* __restrict__ z,
            const float* __restrict__ ln_g, const float* __restrict__ ln_b,
            float* __restrict__ h, float* __restrict__ c)
{
  const int b    = blockIdx.x;
  const int j    = threadIdx.x;
  const int lane = j & 31;
  const int wv   = j >> 5;

  __shared__ float red[16][8];

  const float* zb = z + (unsigned long long)b * GATES;
  float raws[4] = { zb[j], zb[512 + j], zb[1024 + j], zb[1536 + j] };

  float v[8] = { raws[0], raws[1], raws[2], raws[3],
                 raws[0]*raws[0], raws[1]*raws[1], raws[2]*raws[2], raws[3]*raws[3] };
  #pragma unroll
  for (int off = 16; off > 0; off >>= 1)
    #pragma unroll
    for (int i = 0; i < 8; ++i) v[i] += __shfl_down(v[i], off, 32);
  if (lane == 0)
    #pragma unroll
    for (int i = 0; i < 8; ++i) red[wv][i] = v[i];
  __syncthreads();

  float s[8];
  #pragma unroll
  for (int i = 0; i < 8; ++i) {
    float t = 0.f;
    for (int p = 0; p < 16; ++p) t += red[p][i];
    s[i] = t;
  }
  __syncthreads();   // red will be reused below

  const float inv = 1.0f / (float)HDIM;
  float nrm[4];
  #pragma unroll
  for (int g = 0; g < 4; ++g) {
    float m   = s[g] * inv;
    float var = s[4 + g] * inv - m * m;
    nrm[g] = (raws[g] - m) * rsqrtf(var + 1e-5f) * ln_g[g * HDIM + j]
             + ln_b[g * HDIM + j];
  }
  float ig = 1.f / (1.f + __expf(-nrm[0]));
  float fg = 1.f / (1.f + __expf(-nrm[1]));
  float gg = tanhf(nrm[2]);
  float og = 1.f / (1.f + __expf(-nrm[3]));

  float cold = c[(unsigned long long)b * HDIM + j];
  float cnew = fg * cold + ig * gg;

  float w0 = cnew, w1 = cnew * cnew;
  #pragma unroll
  for (int off = 16; off > 0; off >>= 1) {
    w0 += __shfl_down(w0, off, 32);
    w1 += __shfl_down(w1, off, 32);
  }
  if (lane == 0) { red[wv][0] = w0; red[wv][1] = w1; }
  __syncthreads();
  float sc = 0.f, sc2 = 0.f;
  for (int p = 0; p < 16; ++p) { sc += red[p][0]; sc2 += red[p][1]; }

  float mc = sc * inv;
  float vc = sc2 * inv - mc * mc;
  float cn = (cnew - mc) * rsqrtf(vc + 1e-5f) * ln_g[4 * HDIM + j]
             + ln_b[4 * HDIM + j];

  h[(unsigned long long)b * HDIM + j] = og * tanhf(cn);
  c[(unsigned long long)b * HDIM + j] = cnew;
}

// out[b,t,:] = sigmoid(gpre) * h1 + (1-sigmoid(gpre)) * h0
__global__ void __launch_bounds__(512)
highway_out(const float* __restrict__ gpre, const float* __restrict__ h1,
            const float* __restrict__ h0, float* __restrict__ out, int t)
{
  const int b = blockIdx.x;
  const int j = threadIdx.x;
  unsigned long long idx = (unsigned long long)b * HDIM + j;
  float g = 1.f / (1.f + __expf(-gpre[idx]));
  out[((unsigned long long)b * TLEN + t) * HDIM + j] =
      g * h1[idx] + (1.f - g) * h0[idx];
}

__global__ void __launch_bounds__(256)
write_states(const float* h0, const float* h1, const float* c0, const float* c1,
             float* out_tail) {
  int i = blockIdx.x * 256 + threadIdx.x;
  const int n = BATCH * HDIM;   // 32768
  if (i < n) {
    out_tail[i]          = h0[i];   // h_n[0]
    out_tail[n + i]      = h1[i];   // h_n[1]
    out_tail[2 * n + i]  = c0[i];   // c_n[0]
    out_tail[3 * n + i]  = c1[i];   // c_n[1]
  }
}

extern "C" void kernel_launch(void* const* d_in, const int* in_sizes, int n_in,
                              void* d_out, int out_size, void* d_ws, size_t ws_size,
                              hipStream_t stream)
{
  (void)in_sizes; (void)n_in; (void)out_size; (void)ws_size;

  const float* x    = (const float*)d_in[0];
  const float* W0   = (const float*)d_in[1];
  const float* b0   = (const float*)d_in[2];
  const float* lng0 = (const float*)d_in[3];
  const float* lnb0 = (const float*)d_in[4];
  const float* W1   = (const float*)d_in[5];
  const float* b1   = (const float*)d_in[6];
  const float* lng1 = (const float*)d_in[7];
  const float* lnb1 = (const float*)d_in[8];
  const float* hwW  = (const float*)d_in[9];
  const float* hwb  = (const float*)d_in[10];
  float* out = (float*)d_out;

  // Workspace carve-up (256B aligned regions)
  char* ws = (char*)d_ws;
  size_t off = 0;
  auto take = [&](size_t bytes) {
    char* p = ws + off; off += (bytes + 255) & ~(size_t)255; return (void*)p;
  };
  const int nW  = GATES * (DDIM + HDIM);  // 2048*1024
  const int nHW = HDIM * HDIM;            // 512*512
  __bf16* W0hi = (__bf16*)take((size_t)nW * 2);
  __bf16* W0lo = (__bf16*)take((size_t)nW * 2);
  __bf16* W1hi = (__bf16*)take((size_t)nW * 2);
  __bf16* W1lo = (__bf16*)take((size_t)nW * 2);
  __bf16* HWhi = (__bf16*)take((size_t)nHW * 2);
  __bf16* HWlo = (__bf16*)take((size_t)nHW * 2);
  float* z0   = (float*)take((size_t)BATCH * GATES * 4);
  float* z1   = (float*)take((size_t)BATCH * GATES * 4);
  float* gpre = (float*)take((size_t)BATCH * HDIM * 4);
  float* h0   = (float*)take((size_t)BATCH * HDIM * 4);
  float* c0   = (float*)take((size_t)BATCH * HDIM * 4);
  float* h1   = (float*)take((size_t)BATCH * HDIM * 4);
  float* c1   = (float*)take((size_t)BATCH * HDIM * 4);

  // Prologue: split weights to bf16 hi/lo (recomputed every call), zero state.
  split_bf16x2<<<(nW + 255) / 256, 256, 0, stream>>>(W0, W0hi, W0lo, nW);
  split_bf16x2<<<(nW + 255) / 256, 256, 0, stream>>>(W1, W1hi, W1lo, nW);
  split_bf16x2<<<(nHW + 255) / 256, 256, 0, stream>>>(hwW, HWhi, HWlo, nHW);
  zero4<<<(BATCH * HDIM + 255) / 256, 256, 0, stream>>>(h0, c0, h1, c1, BATCH * HDIM);

  const unsigned long long xStride = (unsigned long long)TLEN * DDIM;

  for (int t = 0; t < TLEN; ++t) {
    // Layer 0 gates: concat(x_t, h0_prev) @ W0^T + b0
    gemm_bf16x3<<<GATES / 32, 256, 0, stream>>>(
        x + (unsigned long long)t * DDIM, xStride, DDIM,
        h0, HDIM, W0hi, W0lo, b0, z0, GATES, DDIM + HDIM);
    cell_update<<<BATCH, 512, 0, stream>>>(z0, lng0, lnb0, h0, c0);

    // Layer 1 gates: concat(h0_new, h1_prev) @ W1^T + b1
    gemm_bf16x3<<<GATES / 32, 256, 0, stream>>>(
        h0, HDIM, HDIM, h1, HDIM, W1hi, W1lo, b1, z1, GATES, HDIM + HDIM);
    // Highway gate pre-activation: h0_new @ hwW^T + hwb
    gemm_bf16x3<<<HDIM / 32, 256, 0, stream>>>(
        h0, HDIM, HDIM, h0, HDIM, HWhi, HWlo, hwb, gpre, HDIM, HDIM);
    cell_update<<<BATCH, 512, 0, stream>>>(z1, lng1, lnb1, h1, c1);

    highway_out<<<BATCH, 512, 0, stream>>>(gpre, h1, h0, out, t);
  }

  write_states<<<(BATCH * HDIM + 255) / 256, 256, 0, stream>>>(
      h0, h1, c0, c1, out + (unsigned long long)BATCH * TLEN * HDIM);
}
